// SwinTransformerBlock_4853313044772
// MI455X (gfx1250) — matrix operand order repaired and checked by reading, not compiled, not run on hardware
//
#include <hip/hip_runtime.h>

// ---------------------------------------------------------------------------
// Swin Transformer block for MI455X (gfx1250):
//   bf16 WMMA (f32 accum) GEMMs with TDM (tensor_load_to_lds) double-buffered
//   LDS staging, WMMA attention, fused LN/shift/mask/softmax/GELU epilogues.
// B=16 H=W=64 C=512 WS=8 SS=4 NH=16 HD=32 N=64 windows/img=64 HID=2048
// ---------------------------------------------------------------------------

typedef __bf16 bf16_t;
typedef __attribute__((ext_vector_type(16))) __bf16 v16bf;
typedef __attribute__((ext_vector_type(8)))  __bf16 v8bf;
typedef __attribute__((ext_vector_type(8)))  float  v8f;
typedef __attribute__((ext_vector_type(4)))  unsigned v4u;
typedef __attribute__((ext_vector_type(8)))  int    v8i;
typedef __attribute__((ext_vector_type(4)))  int    v4i;

#define LTOK   65536      // B*H*W tokens
#define CDIM   512
#define QKVDIM 1536
#define HIDDIM 2048
#define NHEAD  16
#define NWIN   1024       // B * 64 windows

#if defined(__has_builtin)
#  if __has_builtin(__builtin_amdgcn_tensor_load_to_lds) && \
      __has_builtin(__builtin_amdgcn_s_wait_tensorcnt)
#    define HAVE_TDM 1
#  endif
#endif
#ifndef HAVE_TDM
#  define HAVE_TDM 0
#endif

__device__ __forceinline__ v16bf cat8(v8bf lo, v8bf hi) {
  return __builtin_shufflevector(lo, hi, 0,1,2,3,4,5,6,7,8,9,10,11,12,13,14,15);
}

__device__ __forceinline__ float wave_sum(float v) {
#pragma unroll
  for (int o = 16; o > 0; o >>= 1) v += __shfl_xor(v, o, 32);
  return v;
}

// region label for the shifted-window mask (H=W=64, WS=8, SS=4)
__device__ __forceinline__ int lab(int p) { return p < 56 ? 0 : (p < 60 ? 1 : 2); }

#if HAVE_TDM
// ---------------------------------------------------------------------------
// TDM: DMA a 2D tile (tile_d0 contiguous bf16 elems per row, tile_d1 rows,
// row pitch = stride elems) from global into LDS at byte offset lds_addr.
// Descriptor layout per CDNA5 ISA (D# groups 0/1).
// ---------------------------------------------------------------------------
__device__ __forceinline__ void tdm_load_2d(unsigned lds_addr, const void* gptr,
                                            unsigned tile_d0, unsigned tile_d1,
                                            unsigned tensor_d0, unsigned tensor_d1,
                                            unsigned long long stride_elems) {
  unsigned long long ga = (unsigned long long)gptr;
  v4u g0;
  g0[0] = 1u;                                             // count=1 (valid), user mode
  g0[1] = lds_addr;                                       // LDS byte address
  g0[2] = (unsigned)(ga & 0xffffffffu);                   // global_addr[31:0]
  g0[3] = (unsigned)((ga >> 32) & 0x01ffffffu) | (2u << 30);  // addr[56:32] | type=2
  v8i g1;
  g1[0] = (int)(1u << 16);                                // data_size=1 (2 bytes)
  g1[1] = (int)((tensor_d0 & 0xffffu) << 16);             // tensor_dim0[15:0]
  g1[2] = (int)((tensor_d0 >> 16) | ((tensor_d1 & 0xffffu) << 16));
  g1[3] = (int)((tensor_d1 >> 16) | (tile_d0 << 16));     // tile_dim0
  g1[4] = (int)(tile_d1 & 0xffffu);                       // tile_dim1 (tile_dim2=0)
  g1[5] = (int)(unsigned)(stride_elems & 0xffffffffu);    // tensor_dim0_stride[31:0]
  g1[6] = (int)(unsigned)((stride_elems >> 32) & 0xffffu);// stride[47:32]
  g1[7] = 0;
  v4i z = {};
#if __clang_major__ >= 23
  v8i z8 = {};
  __builtin_amdgcn_tensor_load_to_lds(g0, g1, z, z, z8, 0);
#else
  __builtin_amdgcn_tensor_load_to_lds(g0, g1, z, z, 0);
#endif
}
#endif

// ---------------------------------------------------------------------------
// fp32 [K][N] weight -> bf16 [N][K] transposed weight
// ---------------------------------------------------------------------------
__global__ __launch_bounds__(256) void transpose_w(const float* __restrict__ W,
                                                   bf16_t* __restrict__ WT,
                                                   int K, int N) {
  int idx = blockIdx.x * 256 + threadIdx.x;
  if (idx >= K * N) return;
  int k = idx / N, n = idx - k * N;
  WT[(size_t)n * K + k] = (bf16_t)W[idx];
}

// biasT[head][n][m] = rpb[rpi(n,m)*NH + head]
__global__ __launch_bounds__(256) void build_bias(const float* __restrict__ rpb,
                                                  float* __restrict__ biasT) {
  int idx = blockIdx.x * 256 + threadIdx.x;       // 16*64*64
  int head = idx >> 12;
  int n = (idx >> 6) & 63, m = idx & 63;
  int i1 = n >> 3, j1 = n & 7, i2 = m >> 3, j2 = m & 7;
  int rp = (i1 - i2 + 7) * 15 + (j1 - j2 + 7);
  biasT[idx] = rpb[rp * NHEAD + head];
}

// ---------------------------------------------------------------------------
// LayerNorm; REMAP=true also does cyclic shift + window partition (LN1),
// writing bf16 rows in window order.  One wave per row, 8 rows per block.
// ---------------------------------------------------------------------------
template <bool REMAP>
__global__ __launch_bounds__(256) void ln_kernel(const float* __restrict__ x,
                                                 const float* __restrict__ g,
                                                 const float* __restrict__ be,
                                                 bf16_t* __restrict__ out) {
  int row  = blockIdx.x * 8 + (threadIdx.x >> 5);
  int lane = threadIdx.x & 31;
  int src  = row;
  if (REMAP) {
    int win = row >> 6, tok = row & 63;
    int b = win >> 6, wi = win & 63;
    int wh = wi >> 3, ww = wi & 7;
    int i = tok >> 3, j = tok & 7;
    int srch = (wh * 8 + i + 4) & 63;   // roll(-SS): shifted pos reads +SS
    int srcw = (ww * 8 + j + 4) & 63;
    src = b * 4096 + srch * 64 + srcw;
  }
  const float* xr = x + (size_t)src * CDIM;
  float vals[16];
  float s = 0.f;
#pragma unroll
  for (int i = 0; i < 16; ++i) { vals[i] = xr[lane + i * 32]; s += vals[i]; }
  float mean = wave_sum(s) * (1.f / CDIM);
  float vv = 0.f;
#pragma unroll
  for (int i = 0; i < 16; ++i) { float d = vals[i] - mean; vv += d * d; }
  float inv = rsqrtf(wave_sum(vv) * (1.f / CDIM) + 1e-5f);
  bf16_t* orow = out + (size_t)row * CDIM;
#pragma unroll
  for (int i = 0; i < 16; ++i) {
    int c = lane + i * 32;
    orow[c] = (bf16_t)((vals[i] - mean) * inv * g[c] + be[c]);
  }
}

// ---------------------------------------------------------------------------
// WMMA GEMM: C[M,N] = A[M,K] (bf16 row-major) * BT[N,K]^T (bf16) + bias.
// 256 threads = 8 waves; wave tile 64x32 (4x2 WMMA tiles), WG tile 128x128.
// A and B 128x32 tiles staged in double-buffered LDS by the Tensor Data
// Mover (wave 0 issues 2 descriptors / k-step, s_wait_tensorcnt<=2 overlap).
// EPI: 0 = bf16 out (QKV)          1 = GELU -> bf16 out (FC1)
//      2 = window-reverse + unshift + residual -> f32 (proj)
//      3 = residual -> f32 (FC2 -> d_out)
// ---------------------------------------------------------------------------
template <int EPI>
__global__ __launch_bounds__(256) void gemm_wmma(const bf16_t* __restrict__ A,
                                                 const bf16_t* __restrict__ BT,
                                                 const float* __restrict__ bias,
                                                 int M, int N, int K,
                                                 bf16_t* __restrict__ outb,
                                                 float* __restrict__ outf,
                                                 const float* __restrict__ res) {
  __shared__ __align__(16) bf16_t ldsA[2][128][32];
  __shared__ __align__(16) bf16_t ldsB[2][128][32];

  int lane  = threadIdx.x & 31;
  int wave  = threadIdx.x >> 5;
  int waveM = wave >> 2;             // 0..1
  int waveN = wave & 3;              // 0..3
  int rowbase = blockIdx.x * 128 + waveM * 64;
  int colbase = blockIdx.y * 128 + waveN * 32;
  int lrow = lane & 15, kg = lane >> 4;

  const bf16_t* Atile = A + (size_t)(blockIdx.x * 128) * K;
  const bf16_t* Btile = BT + (size_t)(blockIdx.y * 128) * K;
  bool issuer = (threadIdx.x < 32);

  v8f acc[4][2];
#pragma unroll
  for (int mt = 0; mt < 4; ++mt)
#pragma unroll
    for (int nt = 0; nt < 2; ++nt) acc[mt][nt] = (v8f){};

  int nsteps = K >> 5;

#if HAVE_TDM
  unsigned offA[2] = { (unsigned)(unsigned long long)&ldsA[0][0][0],
                       (unsigned)(unsigned long long)&ldsA[1][0][0] };
  unsigned offB[2] = { (unsigned)(unsigned long long)&ldsB[0][0][0],
                       (unsigned)(unsigned long long)&ldsB[1][0][0] };
  if (issuer) {
    tdm_load_2d(offA[0], Atile, 32, 128, (unsigned)K, 128, (unsigned)K);
    tdm_load_2d(offB[0], Btile, 32, 128, (unsigned)K, 128, (unsigned)K);
  }
#endif

  for (int i = 0; i < nsteps; ++i) {
    int buf = i & 1;
#if HAVE_TDM
    if (issuer) {
      if (i + 1 < nsteps) {
        int nb = (i + 1) & 1;
        tdm_load_2d(offA[nb], Atile + (i + 1) * 32, 32, 128, (unsigned)K, 128, (unsigned)K);
        tdm_load_2d(offB[nb], Btile + (i + 1) * 32, 32, 128, (unsigned)K, 128, (unsigned)K);
        __builtin_amdgcn_s_wait_tensorcnt(2);   // drain tile i, keep tile i+1 in flight
      } else {
        __builtin_amdgcn_s_wait_tensorcnt(0);
      }
    }
#else
    // fallback: cooperative synchronous copy
    for (int e = threadIdx.x; e < 512; e += 256) {
      int r = e >> 2, c8 = (e & 3) * 8;
      *(v8bf*)&ldsA[buf][r][c8] = *(const v8bf*)&Atile[(size_t)r * K + i * 32 + c8];
      *(v8bf*)&ldsB[buf][r][c8] = *(const v8bf*)&Btile[(size_t)r * K + i * 32 + c8];
    }
#endif
    __syncthreads();   // tile i in LDS, visible to all waves

    v16bf a[4], b[2];
#pragma unroll
    for (int mt = 0; mt < 4; ++mt) {
      const bf16_t* p = &ldsA[buf][waveM * 64 + mt * 16 + lrow][kg * 8];
      a[mt] = cat8(*(const v8bf*)p, *(const v8bf*)(p + 16));
    }
#pragma unroll
    for (int nt = 0; nt < 2; ++nt) {
      const bf16_t* p = &ldsB[buf][waveN * 32 + nt * 16 + lrow][kg * 8];
      b[nt] = cat8(*(const v8bf*)p, *(const v8bf*)(p + 16));
    }
#pragma unroll
    for (int mt = 0; mt < 4; ++mt)
#pragma unroll
      for (int nt = 0; nt < 2; ++nt)
        acc[mt][nt] = __builtin_amdgcn_wmma_f32_16x16x32_bf16(
            false, a[mt], false, b[nt], (short)0, acc[mt][nt], false, false);

    __syncthreads();   // buf reused at step i+2; protect from next issue
  }

#pragma unroll
  for (int mt = 0; mt < 4; ++mt) {
#pragma unroll
    for (int nt = 0; nt < 2; ++nt) {
      int gc = colbase + nt * 16 + lrow;
      float bia = bias[gc];
#pragma unroll
      for (int r = 0; r < 8; ++r) {
        int gr = rowbase + mt * 16 + kg * 8 + r;
        float v = acc[mt][nt][r] + bia;
        if (EPI == 0) {
          outb[(size_t)gr * N + gc] = (bf16_t)v;
        } else if (EPI == 1) {
          float ge = 0.5f * v * (1.0f + erff(v * 0.70710678118f));
          outb[(size_t)gr * N + gc] = (bf16_t)ge;
        } else if (EPI == 2) {
          int win = gr >> 6, tok = gr & 63;
          int b_ = win >> 6, wi = win & 63;
          int wh = wi >> 3, ww = wi & 7;
          int i = tok >> 3, j = tok & 7;
          int h = (wh * 8 + i + 4) & 63;
          int w = (ww * 8 + j + 4) & 63;
          size_t idx = ((size_t)(b_ * 4096 + h * 64 + w)) * CDIM + gc;
          outf[idx] = v + res[idx];
        } else {
          size_t idx = (size_t)gr * N + gc;
          outf[idx] = v + res[idx];
        }
      }
    }
  }
  (void)M;
}

// ---------------------------------------------------------------------------
// Attention: one workgroup per (window, head).  4 waves x 16 query rows.
// S = (Q K^T)*scale + bias + mask ; softmax ; O = P V ; store bf16 head-major.
// ---------------------------------------------------------------------------
__global__ __launch_bounds__(128) void attn_kernel(const bf16_t* __restrict__ qkv,
                                                   const float* __restrict__ biasT,
                                                   bf16_t* __restrict__ obuf) {
  int head = blockIdx.x & 15;
  int win  = blockIdx.x >> 4;
  int lane = threadIdx.x & 31;
  int wave = threadIdx.x >> 5;                 // 0..3
  int lrow = lane & 15, kg = lane >> 4;

  __shared__ __align__(16) bf16_t vT[32][64];           // V^T (ch x key)
  __shared__ float  S[4][16][64];                       // logits per wave
  __shared__ __align__(16) bf16_t P[4][16][64];         // softmax probs bf16

  const size_t rowstride = QKVDIM;
  const bf16_t* qkvw = qkv + (size_t)win * 64 * rowstride;

  // stage V transposed into LDS
  for (int idx = threadIdx.x; idx < 2048; idx += 128) {
    int key = idx >> 5, ch = idx & 31;
    vT[ch][key] = qkvw[(size_t)key * rowstride + (32 + head) * 32 + ch];
  }
  __syncthreads();

  // ---- S = Q K^T (one K-step: HD = 32) --------------------------------
  v16bf aq;
  {
    const bf16_t* p = qkvw + (size_t)(wave * 16 + lrow) * rowstride + head * 32 + kg * 8;
    aq = cat8(*(const v8bf*)p, *(const v8bf*)(p + 16));
  }
  v8f sc[4];
#pragma unroll
  for (int ct = 0; ct < 4; ++ct) {
    const bf16_t* p = qkvw + (size_t)(ct * 16 + lrow) * rowstride + (16 + head) * 32 + kg * 8;
    v16bf bk = cat8(*(const v8bf*)p, *(const v8bf*)(p + 16));
    v8f z = (v8f){};
    sc[ct] = __builtin_amdgcn_wmma_f32_16x16x32_bf16(false, aq, false, bk,
                                                     (short)0, z, false, false);
  }
  const float scale = 0.17677669529663689f;   // HD^-0.5
#pragma unroll
  for (int ct = 0; ct < 4; ++ct)
#pragma unroll
    for (int r = 0; r < 8; ++r)
      S[wave][kg * 8 + r][ct * 16 + lrow] = sc[ct][r] * scale;
  __syncthreads();

  // ---- softmax (+ relative-position bias + shift mask) ----------------
  if (lane < 16) {
    int n  = wave * 16 + lane;                 // query token in window
    int wi = win & 63;
    int wh = wi >> 3, ww = wi & 7;
    int i1 = n >> 3, j1 = n & 7;
    int c1 = lab(wh * 8 + i1) * 3 + lab(ww * 8 + j1);
    const float* brow = biasT + ((size_t)head * 64 + n) * 64;
    float rowv[64];
    float mx = -3.0e38f;
#pragma unroll 8
    for (int m = 0; m < 64; ++m) {
      int i2 = m >> 3, j2 = m & 7;
      int c2 = lab(wh * 8 + i2) * 3 + lab(ww * 8 + j2);
      float t = S[wave][lane][m] + brow[m] + ((c1 != c2) ? -100.0f : 0.0f);
      rowv[m] = t;
      mx = fmaxf(mx, t);
    }
    float sum = 0.f;
#pragma unroll 8
    for (int m = 0; m < 64; ++m) { float e = __expf(rowv[m] - mx); rowv[m] = e; sum += e; }
    float inv = 1.f / sum;
#pragma unroll 8
    for (int m = 0; m < 64; ++m) P[wave][lane][m] = (bf16_t)(rowv[m] * inv);
  }
  __syncthreads();

  // ---- O = P V  (K = 64 keys -> 2 K-steps; N = 32 ch -> 2 tiles) ------
  v8f oc[2];
  oc[0] = (v8f){}; oc[1] = (v8f){};
#pragma unroll
  for (int ks = 0; ks < 2; ++ks) {
    const bf16_t* pa = &P[wave][lrow][ks * 32 + kg * 8];
    v16bf ap = cat8(*(const v8bf*)pa, *(const v8bf*)(pa + 16));
#pragma unroll
    for (int nt = 0; nt < 2; ++nt) {
      const bf16_t* pb = &vT[nt * 16 + lrow][ks * 32 + kg * 8];
      v16bf bv = cat8(*(const v8bf*)pb, *(const v8bf*)(pb + 16));
      oc[nt] = __builtin_amdgcn_wmma_f32_16x16x32_bf16(false, ap, false, bv,
                                                       (short)0, oc[nt], false, false);
    }
  }
#pragma unroll
  for (int nt = 0; nt < 2; ++nt)
#pragma unroll
    for (int r = 0; r < 8; ++r) {
      int tok = wave * 16 + kg * 8 + r;
      int ch  = head * 32 + nt * 16 + lrow;
      obuf[((size_t)win * 64 + tok) * CDIM + ch] = (bf16_t)oc[nt][r];
    }
}

// ---------------------------------------------------------------------------
extern "C" void kernel_launch(void* const* d_in, const int* in_sizes, int n_in,
                              void* d_out, int out_size, void* d_ws, size_t ws_size,
                              hipStream_t stream) {
  const float* x      = (const float*)d_in[0];
  const float* g1     = (const float*)d_in[1];
  const float* b1     = (const float*)d_in[2];
  const float* qkv_w  = (const float*)d_in[3];
  const float* qkv_b  = (const float*)d_in[4];
  const float* proj_w = (const float*)d_in[5];
  const float* proj_b = (const float*)d_in[6];
  const float* rpb    = (const float*)d_in[7];
  const float* g2     = (const float*)d_in[8];
  const float* b2     = (const float*)d_in[9];
  const float* fc1_w  = (const float*)d_in[10];
  const float* fc1_b  = (const float*)d_in[11];
  const float* fc2_w  = (const float*)d_in[12];
  const float* fc2_b  = (const float*)d_in[13];
  float* out = (float*)d_out;

  char* ws = (char*)d_ws;
  // region B (64 MiB): xw -> obuf -> hn (sequential reuse)
  bf16_t* xw   = (bf16_t*)(ws);
  bf16_t* obuf = (bf16_t*)(ws);
  bf16_t* hn   = (bf16_t*)(ws);
  // region A (256 MiB): qkv -> h1
  bf16_t* qkvb = (bf16_t*)(ws + (64ull << 20));
  bf16_t* h1   = (bf16_t*)(ws + (64ull << 20));
  // region X (128 MiB): x after attention residual (fp32)
  float*  xbuf = (float*)(ws + (320ull << 20));
  // weights + bias table (~7 MiB)
  char* wp = ws + (448ull << 20);
  bf16_t* wqkvT = (bf16_t*)(wp);                        // 1536x512
  bf16_t* wprjT = (bf16_t*)(wp + (2ull << 20));         // 512x512
  bf16_t* wfc1T = (bf16_t*)(wp + (3ull << 20));         // 2048x512
  bf16_t* wfc2T = (bf16_t*)(wp + (6ull << 20));         // 512x2048
  float*  biasT = (float*)(wp + (9ull << 20));          // 16x64x64

  // weight transposes (fp32 -> bf16 [N][K])
  transpose_w<<<(CDIM * QKVDIM + 255) / 256, 256, 0, stream>>>(qkv_w, wqkvT, CDIM, QKVDIM);
  transpose_w<<<(CDIM * CDIM + 255) / 256, 256, 0, stream>>>(proj_w, wprjT, CDIM, CDIM);
  transpose_w<<<(CDIM * HIDDIM + 255) / 256, 256, 0, stream>>>(fc1_w, wfc1T, CDIM, HIDDIM);
  transpose_w<<<(HIDDIM * CDIM + 255) / 256, 256, 0, stream>>>(fc2_w, wfc2T, HIDDIM, CDIM);
  build_bias<<<(NHEAD * 64 * 64) / 256, 256, 0, stream>>>(rpb, biasT);

  // LN1 + shift + window partition -> xw (bf16, window order)
  ln_kernel<true><<<LTOK / 8, 256, 0, stream>>>(x, g1, b1, xw);

  // QKV GEMM -> qkvb
  gemm_wmma<0><<<dim3(LTOK / 128, QKVDIM / 128), 256, 0, stream>>>(
      xw, wqkvT, qkv_b, LTOK, QKVDIM, CDIM, qkvb, nullptr, nullptr);

  // attention -> obuf
  attn_kernel<<<NWIN * NHEAD, 128, 0, stream>>>(qkvb, biasT, obuf);

  // proj GEMM + window reverse + unshift + residual -> xbuf (f32)
  gemm_wmma<2><<<dim3(LTOK / 128, CDIM / 128), 256, 0, stream>>>(
      obuf, wprjT, proj_b, LTOK, CDIM, CDIM, nullptr, xbuf, x);

  // LN2 -> hn (bf16)
  ln_kernel<false><<<LTOK / 8, 256, 0, stream>>>(xbuf, g2, b2, hn);

  // FC1 + GELU -> h1 (bf16)
  gemm_wmma<1><<<dim3(LTOK / 128, HIDDIM / 128), 256, 0, stream>>>(
      hn, wfc1T, fc1_b, LTOK, HIDDIM, CDIM, h1, nullptr, nullptr);

  // FC2 + residual -> d_out (f32)
  gemm_wmma<3><<<dim3(LTOK / 128, CDIM / 128), 256, 0, stream>>>(
      h1, wfc2T, fc2_b, LTOK, CDIM, HIDDIM, nullptr, out, xbuf);

  (void)in_sizes; (void)n_in; (void)out_size; (void)ws_size;
}
